// FHEAttention_32066225832521
// MI455X (gfx1250) — matrix operand order repaired and checked by reading, not compiled, not run on hardware
//
#include <hip/hip_runtime.h>
#include <math.h>

// ---------------------------------------------------------------------------
// FHE attention for MI455X (gfx1250, wave32, WMMA f32_16x16x32_f16)
// B=2, S=2048, D=2048, H=16, HD=128
// ---------------------------------------------------------------------------

#define D_MODEL 2048
#define NH      16
#define HD      128
#define BSZ     2
#define SEQ     2048
#define MTOT    (BSZ * SEQ)   // 4096 rows
#define NEG_BIG (-1.0e30f)

typedef __attribute__((ext_vector_type(16))) _Float16 v16h;
typedef __attribute__((ext_vector_type(8)))  float    v8f;

union FragH { v16h v; unsigned u[8]; };
union FragF { v8f  v; float    f[8]; };
union U4H   { uint4 u; _Float16 h[8]; };
union H4U2  { _Float16 h[4]; uint2 u; };

// A-matrix (16x32 f16) per-lane K index for packed pair j (ISA 7.12.2):
// V0..3 hold K={0..7}+8*half, V4..7 hold K={16..23}+8*half, pairs of 2.
__device__ __forceinline__ int kidxA(int j, int half) {
    return ((j >> 2) << 4) + ((j & 3) << 1) + (half << 3);
}

__device__ __forceinline__ float fq(float v) { return rintf(v * 100.0f) * 0.01f; }

// Async global->LDS copy, 16B per lane (GLOBAL_LOAD_ASYNC_TO_LDS_B128,
// ISA 15.18.3 op 98, tracked by ASYNCcnt). LDS dest address is the low
// 32 bits of the generic pointer (LDS aperture: addr[31:0] = LDS offset).
__device__ __forceinline__ void async_ld_b128(_Float16* lds_dst,
                                              const _Float16* gsrc) {
    unsigned ldsa = (unsigned)(unsigned long long)lds_dst;
    unsigned long long ga = (unsigned long long)gsrc;
    asm volatile("global_load_async_to_lds_b128 %0, %1, off"
                 :: "v"(ldsa), "v"(ga) : "memory");
}
__device__ __forceinline__ void wait_async0() {
    asm volatile("s_wait_asynccnt 0x0" ::: "memory");
}
__device__ __forceinline__ void wait_ds0() {
    asm volatile("s_wait_dscnt 0x0" ::: "memory");
}

// ---------------------------------------------------------------------------
// fp32 -> f16 convert (4 elements/thread)
// ---------------------------------------------------------------------------
__global__ void to_half_kernel(const float4* __restrict__ src,
                               uint2* __restrict__ dst, int n4) {
    int i = blockIdx.x * 256 + threadIdx.x;
    if (i < n4) {
        float4 f = src[i];
        H4U2 t;
        t.h[0] = (_Float16)f.x; t.h[1] = (_Float16)f.y;
        t.h[2] = (_Float16)f.z; t.h[3] = (_Float16)f.w;
        dst[i] = t.u;
    }
}

// ---------------------------------------------------------------------------
// C[M,N] = fq(A[M,K] @ W[N,K]^T); A,W f16; out f16 or f32.
// 256 threads = 8 waves; block tile 128x128; wave tile 32x64 (2x4 WMMA C
// tiles); K-step 64; double-buffered async global->LDS pipeline.
// ---------------------------------------------------------------------------
template <bool OUT_F16>
__global__ __launch_bounds__(256) void gemm_fq_kernel(
    const _Float16* __restrict__ A, const _Float16* __restrict__ W,
    _Float16* __restrict__ outH, float* __restrict__ outF,
    int M, int N, int K)
{
    __shared__ _Float16 Xs[2][128 * 64];   // 2 x 16 KB
    __shared__ _Float16 Wt[2][128 * 64];   // 2 x 16 KB

    const int tid  = threadIdx.x;
    const int lane = tid & 31;
    const int wave = tid >> 5;
    const int half = lane >> 4;
    const int ln   = lane & 15;
    const int wm   = wave >> 1;        // wave row (0..3) -> 32 rows each
    const int wn   = wave & 1;         // wave col (0..1) -> 64 cols each
    const int mBlk = blockIdx.y * 128;
    const int nBlk = blockIdx.x * 128;

    FragF c[2][4] = {};

    const int sr  = tid >> 1;           // staging row 0..127
    const int scb = (tid & 1) * 32;     // staging col base (0 / 32)

    auto stage = [&](int buf, int kk) {
        const _Float16* ga = &A[(size_t)(mBlk + sr) * K + kk + scb];
        const _Float16* gw = &W[(size_t)(nBlk + sr) * K + kk + scb];
        #pragma unroll
        for (int rep = 0; rep < 4; rep++) {
            async_ld_b128(&Xs[buf][sr * 64 + scb + rep * 8], ga + rep * 8);
            async_ld_b128(&Wt[buf][sr * 64 + scb + rep * 8], gw + rep * 8);
        }
        if (kk + 128 < K) {             // warm GL2 for tile t+2
            __builtin_prefetch(ga + 128, 0, 1);
            __builtin_prefetch(gw + 128, 0, 1);
        }
    };

    const int nK = K / 64;
    stage(0, 0);
    wait_async0();
    __syncthreads();

    for (int t = 0; t < nK; t++) {
        const int cur = t & 1, nxt = cur ^ 1;
        if (t + 1 < nK) stage(nxt, (t + 1) * 64);   // overlap with compute

        #pragma unroll
        for (int ks = 0; ks < 64; ks += 32) {
            FragH a[2], b[4];
            #pragma unroll
            for (int i = 0; i < 2; i++) {
                const int row = wm * 32 + i * 16 + ln;
                #pragma unroll
                for (int j = 0; j < 8; j++)
                    a[i].u[j] = *(const unsigned*)
                        &Xs[cur][row * 64 + ks + kidxA(j, half)];
            }
            #pragma unroll
            for (int jn = 0; jn < 4; jn++) {
                const int col = wn * 64 + jn * 16 + ln;
                #pragma unroll
                for (int j = 0; j < 8; j++)
                    b[jn].u[j] = *(const unsigned*)
                        &Wt[cur][col * 64 + ks + kidxA(j, half)];
            }
            #pragma unroll
            for (int i = 0; i < 2; i++)
                #pragma unroll
                for (int jn = 0; jn < 4; jn++)
                    c[i][jn].v = __builtin_amdgcn_wmma_f32_16x16x32_f16(
                        false, a[i].v, false, b[jn].v, (short)0, c[i][jn].v,
                        false, false);
        }
        wait_async0();      // next tile landed (own wave's asyncs)
        __syncthreads();    // all waves done reading cur + staging nxt
    }

    // Epilogue: C layout m = r + 8*half, n = ln (ISA 7.12.2)
    #pragma unroll
    for (int i = 0; i < 2; i++)
        #pragma unroll
        for (int jn = 0; jn < 4; jn++)
            #pragma unroll
            for (int r = 0; r < 8; r++) {
                const int row = mBlk + wm * 32 + i * 16 + r + 8 * half;
                const int col = nBlk + wn * 64 + jn * 16 + ln;
                const float v = fq(c[i][jn].f[r]);
                if (OUT_F16) outH[(size_t)row * N + col] = (_Float16)v;
                else         outF[(size_t)row * N + col] = v;
            }
}

// ---------------------------------------------------------------------------
// Causal attention, per (b,h): out = fq(softmax(QK^T/sqrt(HD))) @ V
// grid = (SEQ/64, B*H); 128 threads = 4 waves x 16 queries.
// Pass A: row max/sum (branchless online, per-lane + shfl merge), K tiles
//         double-buffered through async LDS loads.
// Pass B: recompute scores, fq(prob) via per-wave LDS tile, PV WMMAs with
//         V transposed in LDS.
// ---------------------------------------------------------------------------
__global__ __launch_bounds__(128) void attn_kernel(
    const _Float16* __restrict__ Qh, const _Float16* __restrict__ Kh,
    const _Float16* __restrict__ Vh, _Float16* __restrict__ Oh)
{
    __shared__ _Float16 Ks[2][16 * 128];  // 8 KB; pass B views it as 32x128
    __shared__ _Float16 Vts[128 * 32];    // 8 KB, transposed [d][k]
    __shared__ _Float16 Ps[4][16 * 32];   // 4 KB, per-wave prob tile

    const int tid  = threadIdx.x;
    const int lane = tid & 31;
    const int wave = tid >> 5;
    const int half = lane >> 4;
    const int ln   = lane & 15;
    const int qt   = blockIdx.x;
    const int bh   = blockIdx.y;
    const int bb   = bh / NH;
    const int hh   = bh % NH;
    const size_t rowBase = (size_t)bb * SEQ;
    const int colBase = hh * HD;
    const int q0 = qt * 64 + wave * 16;
    const float scale = 0.08838834764831845f;   // 1/sqrt(128)

    // Q fragments (4 chunks of K=32), kept in registers for both passes
    FragH qf[4];
    {
        const size_t qrow = (rowBase + q0 + ln) * D_MODEL + colBase;
        #pragma unroll
        for (int dc = 0; dc < 4; dc++)
            #pragma unroll
            for (int j = 0; j < 8; j++)
                qf[dc].u[j] = *(const unsigned*)&Qh[qrow + dc * 32 + kidxA(j, half)];
    }

    float mrow[8], srow[8];
    #pragma unroll
    for (int r = 0; r < 8; r++) { mrow[r] = NEG_BIG; srow[r] = 0.0f; }

    // ---------------- Pass A: row max & sum of exp ----------------
    auto stageK16 = [&](int buf, int key0) {
        const int r = tid >> 3, cc = (tid & 7) * 16;
        const _Float16* g = &Kh[(rowBase + key0 + r) * D_MODEL + colBase + cc];
        async_ld_b128(&Ks[buf][r * 128 + cc],     g);
        async_ld_b128(&Ks[buf][r * 128 + cc + 8], g + 8);
    };

    const int nc16 = (qt + 1) * 4;
    stageK16(0, 0);
    wait_async0();
    __syncthreads();

    for (int c = 0; c < nc16; c++) {
        const int cur = c & 1, nxt = cur ^ 1;
        if (c + 1 < nc16) stageK16(nxt, (c + 1) * 16);   // overlap

        FragF sc = {};
        #pragma unroll
        for (int dc = 0; dc < 4; dc++) {
            FragH bk;   // B[d][k] = K[k][d]: pairs over d contiguous in Ks rows
            #pragma unroll
            for (int j = 0; j < 8; j++)
                bk.u[j] = *(const unsigned*)
                    &Ks[cur][ln * 128 + dc * 32 + kidxA(j, half)];
            sc.v = __builtin_amdgcn_wmma_f32_16x16x32_f16(
                false, qf[dc].v, false, bk.v, (short)0, sc.v, false, false);
        }
        const int key = c * 16 + ln;
        #pragma unroll
        for (int r = 0; r < 8; r++) {
            const int q = q0 + r + 8 * half;
            // branchless masked online update (mrow stays finite >= NEG_BIG)
            const float s  = (key <= q) ? sc.f[r] * scale : -INFINITY;
            const float mo = mrow[r];
            const float mn = fmaxf(mo, s);
            srow[r] = srow[r] * __expf(mo - mn) + __expf(s - mn);
            mrow[r] = mn;
        }
        wait_async0();
        __syncthreads();
    }

    // Cross-lane logsumexp merge over the 16 columns of each half-wave
    #pragma unroll
    for (int r = 0; r < 8; r++) {
        float m = mrow[r], s = srow[r];
        for (int d = 1; d < 16; d <<= 1) {
            const float mo = __shfl_xor(m, d, 32);
            const float so = __shfl_xor(s, d, 32);
            const float mn = fmaxf(m, mo);
            s = s * __expf(m - mn) + so * __expf(mo - mn);
            m = mn;
        }
        mrow[r] = m;
        srow[r] = 1.0f / s;            // reciprocal of row sum (s >= 1)
    }

    // ---------------- Pass B: probs -> fq -> PV ----------------
    _Float16* KsFlat = &Ks[0][0];      // 32 x 128 view
    FragF o[8] = {};
    const int nc32 = (qt + 1) * 2;
    for (int c = 0; c < nc32; c++) {
        const int key0 = c * 32;
        {
            const int r = tid >> 2, cc = (tid & 3) * 32;
            const _Float16* g  = &Kh[(rowBase + key0 + r) * D_MODEL + colBase + cc];
            const _Float16* gv = &Vh[(rowBase + key0 + r) * D_MODEL + colBase + cc];
            #pragma unroll
            for (int rep = 0; rep < 4; rep++)
                async_ld_b128(&KsFlat[r * 128 + cc + rep * 8], g + rep * 8);
            #pragma unroll
            for (int rep = 0; rep < 4; rep++) {
                U4H t; t.u = *(const uint4*)(gv + rep * 8);
                #pragma unroll
                for (int e = 0; e < 8; e++)
                    Vts[(cc + rep * 8 + e) * 32 + r] = t.h[e];   // [d][k]
            }
        }
        wait_async0();
        __syncthreads();

        #pragma unroll
        for (int sub = 0; sub < 2; sub++) {
            FragF sc = {};
            #pragma unroll
            for (int dc = 0; dc < 4; dc++) {
                FragH bk;
                #pragma unroll
                for (int j = 0; j < 8; j++)
                    bk.u[j] = *(const unsigned*)&KsFlat[(sub * 16 + ln) * 128 +
                                                        dc * 32 + kidxA(j, half)];
                sc.v = __builtin_amdgcn_wmma_f32_16x16x32_f16(
                    false, qf[dc].v, false, bk.v, (short)0, sc.v, false, false);
            }
            const int key = key0 + sub * 16 + ln;
            #pragma unroll
            for (int r = 0; r < 8; r++) {
                const int q = q0 + r + 8 * half;
                const float e = __expf(sc.f[r] * scale - mrow[r]) * srow[r];
                const float p = (key <= q) ? fq(e) : 0.0f;   // branchless mask
                Ps[wave][(r + 8 * half) * 32 + sub * 16 + ln] = (_Float16)p;
            }
        }
        // Own-wave LDS RAW: prob stores must land before frag loads
        wait_ds0();

        FragH pa;   // A = prob tile 16x32
        #pragma unroll
        for (int j = 0; j < 8; j++)
            pa.u[j] = *(const unsigned*)&Ps[wave][ln * 32 + kidxA(j, half)];

        #pragma unroll
        for (int nt = 0; nt < 8; nt++) {
            FragH bv;   // B[k][d] from transposed Vts: pairs over k contiguous
            #pragma unroll
            for (int j = 0; j < 8; j++)
                bv.u[j] = *(const unsigned*)&Vts[(nt * 16 + ln) * 32 +
                                                 kidxA(j, half)];
            o[nt].v = __builtin_amdgcn_wmma_f32_16x16x32_f16(
                false, pa.v, false, bv.v, (short)0, o[nt].v, false, false);
        }
        __syncthreads();
    }

    #pragma unroll
    for (int nt = 0; nt < 8; nt++)
        #pragma unroll
        for (int r = 0; r < 8; r++) {
            const int q = q0 + r + 8 * half;
            Oh[(rowBase + q) * D_MODEL + colBase + nt * 16 + ln] = (_Float16)o[nt].f[r];
        }
}

// ---------------------------------------------------------------------------
// Launcher
// ---------------------------------------------------------------------------
extern "C" void kernel_launch(void* const* d_in, const int* in_sizes, int n_in,
                              void* d_out, int out_size, void* d_ws, size_t ws_size,
                              hipStream_t stream)
{
    const float* x  = (const float*)d_in[0];
    const float* Wq = (const float*)d_in[1];
    const float* Wk = (const float*)d_in[2];
    const float* Wv = (const float*)d_in[3];
    const float* Wo = (const float*)d_in[4];
    float* out = (float*)d_out;

    const size_t XE = (size_t)MTOT * D_MODEL;      // 8,388,608 elements
    const size_t WE = (size_t)D_MODEL * D_MODEL;   // 4,194,304 elements

    char* ws = (char*)d_ws;
    _Float16* xh  = (_Float16*)ws; ws += XE * 2;
    _Float16* wqh = (_Float16*)ws; ws += WE * 2;
    _Float16* wkh = (_Float16*)ws; ws += WE * 2;
    _Float16* wvh = (_Float16*)ws; ws += WE * 2;
    _Float16* woh = (_Float16*)ws; ws += WE * 2;
    _Float16* qh  = (_Float16*)ws; ws += XE * 2;
    _Float16* kh  = (_Float16*)ws; ws += XE * 2;
    _Float16* vh  = (_Float16*)ws; ws += XE * 2;
    _Float16* oh  = (_Float16*)ws; ws += XE * 2;

    const int X4 = (int)(XE / 4), W4 = (int)(WE / 4);
    to_half_kernel<<<(X4 + 255) / 256, 256, 0, stream>>>(
        (const float4*)x, (uint2*)xh, X4);
    to_half_kernel<<<(W4 + 255) / 256, 256, 0, stream>>>(
        (const float4*)Wq, (uint2*)wqh, W4);
    to_half_kernel<<<(W4 + 255) / 256, 256, 0, stream>>>(
        (const float4*)Wk, (uint2*)wkh, W4);
    to_half_kernel<<<(W4 + 255) / 256, 256, 0, stream>>>(
        (const float4*)Wv, (uint2*)wvh, W4);
    to_half_kernel<<<(W4 + 255) / 256, 256, 0, stream>>>(
        (const float4*)Wo, (uint2*)woh, W4);

    dim3 gg(D_MODEL / 128, MTOT / 128);   // (16, 32)
    gemm_fq_kernel<true><<<gg, 256, 0, stream>>>(xh, wqh, qh, nullptr,
                                                 MTOT, D_MODEL, D_MODEL);
    gemm_fq_kernel<true><<<gg, 256, 0, stream>>>(xh, wkh, kh, nullptr,
                                                 MTOT, D_MODEL, D_MODEL);
    gemm_fq_kernel<true><<<gg, 256, 0, stream>>>(xh, wvh, vh, nullptr,
                                                 MTOT, D_MODEL, D_MODEL);

    dim3 ga(SEQ / 64, BSZ * NH);          // (32, 32)
    attn_kernel<<<ga, 128, 0, stream>>>(qh, kh, vh, oh);

    gemm_fq_kernel<false><<<gg, 256, 0, stream>>>(oh, woh, nullptr, out,
                                                  MTOT, D_MODEL, D_MODEL);
}
